// MultiHeadAttention_26362509263110
// MI455X (gfx1250) — matrix operand order repaired and checked by reading, not compiled, not run on hardware
//
#include <hip/hip_runtime.h>
#include <hip/hip_bf16.h>
#include <stdint.h>

// ---------------------------------------------------------------------------
// MultiHeadAttention (feature-dim attention) + residual + LayerNorm, MI455X.
// bf16 WMMA (v_wmma_f32_16x16x32_bf16) for all GEMMs, f32 accumulation.
// Double-buffered LDS staging via GLOBAL_LOAD_ASYNC_TO_LDS_B128 + ASYNCcnt.
// ---------------------------------------------------------------------------

typedef __bf16 bf16_t;
typedef __attribute__((ext_vector_type(16))) __bf16 v16bf;
typedef __attribute__((ext_vector_type(8)))  __bf16 v8bf;
typedef __attribute__((ext_vector_type(8)))  float  v8f;

#define B_  8
#define S_  4096
#define D_  1024
#define H_  16
#define DK_ 64

// ---- CDNA5 async memory->LDS (bypasses VGPRs, tracked by ASYNCcnt) --------
__device__ __forceinline__ void async_load_b128(uint32_t lds_byte_addr,
                                                const void* gaddr) {
  asm volatile("global_load_async_to_lds_b128 %0, %1, off"
               :: "v"(lds_byte_addr), "v"(gaddr)
               : "memory");
}

template <int N>
__device__ __forceinline__ void wait_asynccnt() {
  asm volatile("s_wait_asynccnt %0" :: "i"(N) : "memory");
}

__device__ __forceinline__ uint32_t lds_addr(const void* p) {
  return (uint32_t)(uintptr_t)p;   // addrspace(3) pointers are LDS byte offsets
}

// ---------------------------------------------------------------------------
__device__ __forceinline__ v8f zero_v8f() {
  v8f z;
#pragma unroll
  for (int i = 0; i < 8; ++i) z[i] = 0.0f;
  return z;
}

__device__ __forceinline__ v16bf frag_from_halves(v8bf lo, v8bf hi) {
  v16bf f;
#pragma unroll
  for (int i = 0; i < 8; ++i) { f[i] = lo[i]; f[i + 8] = hi[i]; }
  return f;
}

// A-fragment (16Mx32K bf16) from LDS tile stored row-major [m][k].
// Lane L (L<16): m=m0+L, k in {0..7, 16..23}; lane L>=16: k in {8..15, 24..31}.
__device__ __forceinline__ v16bf load_a_frag(const bf16_t* lds, int stride,
                                             int m0, int k0) {
  const int lane = threadIdx.x & 31;
  const int m  = m0 + (lane & 15);
  const int kb = (lane < 16) ? 0 : 8;
  const bf16_t* p = lds + m * stride + k0 + kb;
  v8bf lo = *(const v8bf*)(p);        // k = kb .. kb+7
  v8bf hi = *(const v8bf*)(p + 16);   // k = kb+16 .. kb+23
  return frag_from_halves(lo, hi);
}

// B-fragment (32Kx16N bf16) from LDS tile stored K-major per column: [n][k].
// Lane L: n = n0 + L%16, k in {0..7, 16..23} (L<16) or {8..15, 24..31} (L>=16).
__device__ __forceinline__ v16bf load_b_frag(const bf16_t* ldsT, int stride,
                                             int n0, int k0) {
  const int lane = threadIdx.x & 31;
  const int n  = n0 + (lane & 15);
  const int kb = (lane < 16) ? 0 : 8;
  const bf16_t* p = ldsT + n * stride + k0 + kb;
  v8bf lo = *(const v8bf*)(p);
  v8bf hi = *(const v8bf*)(p + 16);
  return frag_from_halves(lo, hi);
}

__device__ __forceinline__ v8f wmma_bf16(v16bf a, v16bf b, v8f c) {
  // (neg_a, A, neg_b, B, c_mod, C, reuse_a, reuse_b)
  return __builtin_amdgcn_wmma_f32_16x16x32_bf16(false, a, false, b,
                                                 (short)0, c, false, false);
}

// ---------------------------------------------------------------------------
// Activation convert: f32 -> bf16 (straight copy), 4 elems/thread.
// ---------------------------------------------------------------------------
__global__ __launch_bounds__(256)
void athena_xconv(const float* __restrict__ x, bf16_t* __restrict__ xb) {
  const size_t i = ((size_t)blockIdx.x * 256 + threadIdx.x) * 4;
  const float4 f = *(const float4*)(x + i);
  bf16_t* o = xb + i;
  o[0] = (bf16_t)f.x; o[1] = (bf16_t)f.y;
  o[2] = (bf16_t)f.z; o[3] = (bf16_t)f.w;
}

// ---------------------------------------------------------------------------
// Weight convert + transpose:  W[k][n] f32  ->  Wt[n][k] bf16   (1024x1024)
// ---------------------------------------------------------------------------
__global__ __launch_bounds__(256)
void athena_wconv(const float* __restrict__ w, bf16_t* __restrict__ wt) {
  const int idx = blockIdx.x * 256 + threadIdx.x;   // idx = n*1024 + k
  const int n = idx >> 10;
  const int k = idx & 1023;
  wt[idx] = (bf16_t)w[k * 1024 + n];
}

// ---------------------------------------------------------------------------
// Generic projection GEMM: C = A[32768,1024](bf16) @ B[1024,1024](bf16,[n][k])
//   Block tile 128x128, BK=32, 256 threads (8 waves, each 2x4 WMMA tiles).
//   Double-buffered async LDS staging (4 async b128 per thread per tile).
// Epilogues:
//   EPI_HEADT    : out bf16 [b][h][e][s]  (per-lane 16B contiguous stores)
//   EPI_NAT_BF16 : out bf16 [m][n]
//   EPI_NAT_F32  : out f32  [m][n]
// ---------------------------------------------------------------------------
enum { EPI_HEADT = 0, EPI_NAT_BF16 = 1, EPI_NAT_F32 = 2 };

template <int EPI>
__global__ __launch_bounds__(256)
void athena_gemm_bb(const bf16_t* __restrict__ A, const bf16_t* __restrict__ Bt,
                    void* __restrict__ Outv, float alpha) {
  __shared__ bf16_t sA[2][128 * 40];   // [m][k], padded stride
  __shared__ bf16_t sB[2][128 * 40];   // [n][k], padded stride

  const int t    = threadIdx.x;
  const int mblk = blockIdx.x;      // 0..255
  const int nblk = blockIdx.y;      // 0..7
  const int wave = t >> 5;
  const int lane = t & 31;
  const int wm   = wave >> 1;       // 0..3  (32 rows each)
  const int wn   = wave & 1;        // 0..1  (64 cols each)

  v8f zf = zero_v8f();
  v8f acc[2][4];
#pragma unroll
  for (int i = 0; i < 2; ++i)
#pragma unroll
    for (int j = 0; j < 4; ++j) acc[i][j] = zf;

  const int row  = t >> 1;          // 0..127
  const int half = t & 1;           // 16-element half of the 32-wide k slice
  const bf16_t* gA = A  + (size_t)(mblk * 128 + row) * 1024 + half * 16;
  const bf16_t* gB = Bt + (size_t)(nblk * 128 + row) * 1024 + half * 16;
  const int ldsOff = row * 40 + half * 16;

  auto issue_tile = [&](int kt, int buf) {
    const bf16_t* ga = gA + kt * 32;
    const bf16_t* gb = gB + kt * 32;
    const uint32_t la = lds_addr(&sA[buf][ldsOff]);
    const uint32_t lb = lds_addr(&sB[buf][ldsOff]);
    async_load_b128(la,      ga);
    async_load_b128(la + 16, ga + 8);
    async_load_b128(lb,      gb);
    async_load_b128(lb + 16, gb + 8);
  };

  issue_tile(0, 0);
  for (int kt = 0; kt < 32; ++kt) {
    const int buf = kt & 1;
    if (kt < 31) {
      issue_tile(kt + 1, buf ^ 1);
      wait_asynccnt<4>();            // current tile landed, next in flight
    } else {
      wait_asynccnt<0>();
    }
    __syncthreads();
    // ---- WMMA compute ----
    v16bf afr[2], bfr[4];
#pragma unroll
    for (int mi = 0; mi < 2; ++mi)
      afr[mi] = load_a_frag(sA[buf], 40, wm * 32 + mi * 16, 0);
#pragma unroll
    for (int ni = 0; ni < 4; ++ni)
      bfr[ni] = load_b_frag(sB[buf], 40, wn * 64 + ni * 16, 0);
#pragma unroll
    for (int mi = 0; mi < 2; ++mi)
#pragma unroll
      for (int ni = 0; ni < 4; ++ni)
        acc[mi][ni] = wmma_bf16(afr[mi], bfr[ni], acc[mi][ni]);
    __syncthreads();                 // done reading buf before it is reloaded
  }

  // ---- epilogue ----
  const int lh = lane >> 4;   // 0/1  -> C rows m0+r or m0+8+r
  const int ln = lane & 15;   // column within tile
#pragma unroll
  for (int mi = 0; mi < 2; ++mi) {
#pragma unroll
    for (int ni = 0; ni < 4; ++ni) {
      const int mbase = mblk * 128 + wm * 32 + mi * 16 + lh * 8;
      const int n     = nblk * 128 + wn * 64 + ni * 16 + ln;
      if (EPI == EPI_HEADT) {
        const int b = mbase >> 12;           // m = b*S + s
        const int s = mbase & (S_ - 1);
        const int h = n >> 6;
        const int e = n & 63;
        bf16_t* out = (bf16_t*)Outv +
                      ((size_t)((b * H_ + h) * DK_ + e)) * S_ + s;
        v8bf w;
#pragma unroll
        for (int r = 0; r < 8; ++r) w[r] = (bf16_t)(acc[mi][ni][r] * alpha);
        *(v8bf*)out = w;                     // 8 consecutive s -> 16B store
      } else if (EPI == EPI_NAT_BF16) {
        bf16_t* out = (bf16_t*)Outv;
#pragma unroll
        for (int r = 0; r < 8; ++r)
          out[(size_t)(mbase + r) * 1024 + n] = (bf16_t)(acc[mi][ni][r] * alpha);
      } else {
        float* out = (float*)Outv;
#pragma unroll
        for (int r = 0; r < 8; ++r)
          out[(size_t)(mbase + r) * 1024 + n] = acc[mi][ni][r] * alpha;
      }
    }
  }
}

// ---------------------------------------------------------------------------
// scores[b,h] = (Q_h^T K_h)  -- 64x64 output, contraction over S=4096.
// Q_t/K_t stored [b][h][d][s] bf16 (s contiguous) so both fragment loads are
// contiguous along the contraction dim. Q already carries the 1/sqrt(DK) scale.
// One workgroup (4 waves) per (b,h); wave w owns m-tile w x 4 n-tiles.
// Double-buffered async staging hides HBM latency over the 128 K-steps.
// ---------------------------------------------------------------------------
__global__ __launch_bounds__(128)
void athena_scores(const bf16_t* __restrict__ qt, const bf16_t* __restrict__ kt,
                   float* __restrict__ scores) {
  __shared__ bf16_t sQ[2][64 * 40];
  __shared__ bf16_t sK[2][64 * 40];
  const int bh   = blockIdx.x;            // 0..127
  const int t    = threadIdx.x;
  const int wave = t >> 5;
  const int lane = t & 31;
  const int row  = t >> 1;                // 0..63
  const int half = t & 1;
  const bf16_t* gq = qt + (size_t)bh * DK_ * S_ + (size_t)row * S_ + half * 16;
  const bf16_t* gk = kt + (size_t)bh * DK_ * S_ + (size_t)row * S_ + half * 16;
  const int ldsOff = row * 40 + half * 16;

  auto issue_tile = [&](int kb, int buf) {
    const bf16_t* q = gq + kb * 32;
    const bf16_t* k = gk + kb * 32;
    const uint32_t lq = lds_addr(&sQ[buf][ldsOff]);
    const uint32_t lk = lds_addr(&sK[buf][ldsOff]);
    async_load_b128(lq,      q);
    async_load_b128(lq + 16, q + 8);
    async_load_b128(lk,      k);
    async_load_b128(lk + 16, k + 8);
  };

  v8f acc[4];
  v8f zf = zero_v8f();
#pragma unroll
  for (int i = 0; i < 4; ++i) acc[i] = zf;

  issue_tile(0, 0);
  for (int kb = 0; kb < S_ / 32; ++kb) {
    const int buf = kb & 1;
    if (kb < S_ / 32 - 1) {
      issue_tile(kb + 1, buf ^ 1);
      wait_asynccnt<4>();
    } else {
      wait_asynccnt<0>();
    }
    __syncthreads();
    v16bf a = load_a_frag(sQ[buf], 40, wave * 16, 0);
#pragma unroll
    for (int ni = 0; ni < 4; ++ni) {
      v16bf b = load_b_frag(sK[buf], 40, ni * 16, 0);
      acc[ni] = wmma_bf16(a, b, acc[ni]);
    }
    __syncthreads();
  }

  const int lh = lane >> 4, ln = lane & 15;
#pragma unroll
  for (int ni = 0; ni < 4; ++ni) {
    const int d0 = wave * 16 + lh * 8;
    const int e  = ni * 16 + ln;
#pragma unroll
    for (int r = 0; r < 8; ++r)
      scores[(size_t)bh * 4096 + (size_t)(d0 + r) * 64 + e] = acc[ni][r];
  }
}

// ---------------------------------------------------------------------------
// softmax over e (last dim of 64) per row d; writes bf16 attn [b][h][d][e].
// ---------------------------------------------------------------------------
__global__ __launch_bounds__(64)
void athena_softmax(const float* __restrict__ scores, bf16_t* __restrict__ attn) {
  const int bh = blockIdx.x;
  const int d  = threadIdx.x;             // 0..63
  const float* row = scores + ((size_t)bh * 64 + d) * 64;
  float mx = -3.4e38f;
#pragma unroll
  for (int e = 0; e < 64; ++e) mx = fmaxf(mx, row[e]);
  float ex[64];
  float sum = 0.0f;
#pragma unroll
  for (int e = 0; e < 64; ++e) { ex[e] = __expf(row[e] - mx); sum += ex[e]; }
  const float inv = 1.0f / sum;
  bf16_t* o = attn + ((size_t)bh * 64 + d) * 64;
#pragma unroll
  for (int e = 0; e < 64; ++e) o[e] = (bf16_t)(ex[e] * inv);
}

// ---------------------------------------------------------------------------
// context^T[s,d] = sum_e V[s,e] * attn[d,e]   per (b,h)
//   GEMM: M=4096(s), N=64(d), K=64(e). A = V natural (k=e contiguous),
//   B = attn natural [d][e] == exactly the LDS [n][k] layout. Output lands
//   directly in ctx bf16 [b][s][h*64+d] — the layout the O-projection wants.
// Grid (32 s-chunks, H, B); 8 waves, wave w -> 16 s rows x 4 d tiles.
// ---------------------------------------------------------------------------
__global__ __launch_bounds__(256)
void athena_context(const bf16_t* __restrict__ v, const bf16_t* __restrict__ attn,
                    bf16_t* __restrict__ ctx) {
  __shared__ bf16_t sV[128 * 72];   // [s_local][e]
  __shared__ bf16_t sA[64 * 72];    // [d][e]  (B as [n][k])
  const int c = blockIdx.x;         // s-chunk, 0..31
  const int h = blockIdx.y;
  const int b = blockIdx.z;
  const int t = threadIdx.x;
  const int wave = t >> 5, lane = t & 31;
  const int s0 = c * 128;

  // stage V tile (128 x 64) and attn (64 x 64) via async copies
#pragma unroll
  for (int p = t; p < 512; p += 256) {
    const int r = p >> 2, q = p & 3;
    async_load_b128(lds_addr(&sV[r * 72 + q * 16]),
                    v + ((size_t)(b * S_ + s0 + r)) * 1024 + h * 64 + q * 16);
  }
  {
    const int r = t >> 2, q = t & 3;
    async_load_b128(lds_addr(&sA[r * 72 + q * 16]),
                    attn + ((size_t)(b * H_ + h) * 64 + r) * 64 + q * 16);
  }
  wait_asynccnt<0>();
  __syncthreads();

  v8f acc[4];
  v8f zf = zero_v8f();
#pragma unroll
  for (int i = 0; i < 4; ++i) acc[i] = zf;

#pragma unroll
  for (int ks = 0; ks < 2; ++ks) {
    v16bf a = load_a_frag(sV, 72, wave * 16, ks * 32);
#pragma unroll
    for (int ni = 0; ni < 4; ++ni) {
      v16bf bf = load_b_frag(sA, 72, ni * 16, ks * 32);
      acc[ni] = wmma_bf16(a, bf, acc[ni]);
    }
  }

  const int lh = lane >> 4, ln = lane & 15;
#pragma unroll
  for (int ni = 0; ni < 4; ++ni) {
    const int sl = wave * 16 + lh * 8;
    const int d  = ni * 16 + ln;
#pragma unroll
    for (int r = 0; r < 8; ++r)
      ctx[((size_t)(b * S_ + s0 + sl + r)) * 1024 + h * 64 + d] =
          (bf16_t)(acc[ni][r]);
  }
}

// ---------------------------------------------------------------------------
// out = LayerNorm(proj + residual), one block (256 thr) per row of 1024.
// ---------------------------------------------------------------------------
__global__ __launch_bounds__(256)
void athena_layernorm(const float* __restrict__ proj,
                      const float* __restrict__ resid,
                      float* __restrict__ out) {
  const int row = blockIdx.x;
  const int t   = threadIdx.x;
  const float* p = proj  + (size_t)row * 1024;
  const float* r = resid + (size_t)row * 1024;
  float x[4];
  float s1 = 0.0f, s2 = 0.0f;
#pragma unroll
  for (int j = 0; j < 4; ++j) {
    const float xv = p[t + 256 * j] + r[t + 256 * j];
    x[j] = xv; s1 += xv; s2 += xv * xv;
  }
  // wave32 reduction
#pragma unroll
  for (int off = 16; off >= 1; off >>= 1) {
    s1 += __shfl_xor(s1, off);
    s2 += __shfl_xor(s2, off);
  }
  __shared__ float w1[8], w2[8];
  if ((t & 31) == 0) { w1[t >> 5] = s1; w2[t >> 5] = s2; }
  __syncthreads();
  float S1 = 0.0f, S2 = 0.0f;
#pragma unroll
  for (int k = 0; k < 8; ++k) { S1 += w1[k]; S2 += w2[k]; }
  const float mean = S1 * (1.0f / 1024.0f);
  const float var  = S2 * (1.0f / 1024.0f) - mean * mean;
  const float inv  = rsqrtf(var + 1e-5f);
  float* o = out + (size_t)row * 1024;
#pragma unroll
  for (int j = 0; j < 4; ++j) o[t + 256 * j] = (x[j] - mean) * inv;
}

// ---------------------------------------------------------------------------
// Launcher
// ---------------------------------------------------------------------------
extern "C" void kernel_launch(void* const* d_in, const int* in_sizes, int n_in,
                              void* d_out, int out_size, void* d_ws, size_t ws_size,
                              hipStream_t stream) {
  (void)in_sizes; (void)n_in; (void)out_size; (void)ws_size;
  const float* inQ = (const float*)d_in[0];
  const float* inK = (const float*)d_in[1];
  const float* inV = (const float*)d_in[2];
  const float* wQ  = (const float*)d_in[3];
  const float* wK  = (const float*)d_in[4];
  const float* wV  = (const float*)d_in[5];
  const float* wO  = (const float*)d_in[6];
  float* out = (float*)d_out;

  char* ws = (char*)d_ws;
  const size_t MB = 1ull << 20;
  bf16_t* wq_t   = (bf16_t*)(ws + 0 * MB);    //   2 MB
  bf16_t* wk_t   = (bf16_t*)(ws + 2 * MB);    //   2 MB
  bf16_t* wv_t   = (bf16_t*)(ws + 4 * MB);    //   2 MB
  bf16_t* wo_t   = (bf16_t*)(ws + 6 * MB);    //   2 MB
  bf16_t* q_t    = (bf16_t*)(ws + 8 * MB);    //  64 MB  [B][H][64][S]
  bf16_t* k_t    = (bf16_t*)(ws + 72 * MB);   //  64 MB  [B][H][64][S]
  bf16_t* v_n    = (bf16_t*)(ws + 136 * MB);  //  64 MB  [B*S][1024]
  float*  scores = (float*) (ws + 200 * MB);  //   2 MB  [B][H][64][64]
  bf16_t* attn   = (bf16_t*)(ws + 202 * MB);  //   1 MB
  bf16_t* xq     = (bf16_t*)(ws + 204 * MB);  //  64 MB  (later reused as ctx)
  bf16_t* xk     = (bf16_t*)(ws + 268 * MB);  //  64 MB  (later reused as proj lo)
  bf16_t* xv     = (bf16_t*)(ws + 332 * MB);  //  64 MB  (later reused as proj hi)
  bf16_t* ctx    = xq;                        //  alias: xq dead after Q-proj
  float*  proj   = (float*)(ws + 268 * MB);   //  128 MB alias: xk/xv dead by then

  const int XBLK = (B_ * S_ * D_) / (256 * 4);   // 32768 blocks

  // 1) activations -> bf16; weights -> bf16 transposed [n][k]
  athena_xconv<<<XBLK, 256, 0, stream>>>(inQ, xq);
  athena_xconv<<<XBLK, 256, 0, stream>>>(inK, xk);
  athena_xconv<<<XBLK, 256, 0, stream>>>(inV, xv);
  athena_wconv<<<4096, 256, 0, stream>>>(wQ, wq_t);
  athena_wconv<<<4096, 256, 0, stream>>>(wK, wk_t);
  athena_wconv<<<4096, 256, 0, stream>>>(wV, wv_t);
  athena_wconv<<<4096, 256, 0, stream>>>(wO, wo_t);

  // 2) projections (Q pre-scaled by 1/sqrt(DK)=0.125)
  dim3 gp(256, 8);
  athena_gemm_bb<EPI_HEADT   ><<<gp, 256, 0, stream>>>(xq, wq_t, q_t, 0.125f);
  athena_gemm_bb<EPI_HEADT   ><<<gp, 256, 0, stream>>>(xk, wk_t, k_t, 1.0f);
  athena_gemm_bb<EPI_NAT_BF16><<<gp, 256, 0, stream>>>(xv, wv_t, v_n, 1.0f);

  // 3) feature-dim attention
  athena_scores <<<B_ * H_, 128, 0, stream>>>(q_t, k_t, scores);
  athena_softmax<<<B_ * H_, 64, 0, stream>>>(scores, attn);
  athena_context<<<dim3(32, H_, B_), 256, 0, stream>>>(v_n, attn, ctx);

  // 4) output projection (f32 out) + fused residual/LayerNorm
  athena_gemm_bb<EPI_NAT_F32><<<gp, 256, 0, stream>>>(ctx, wo_t, proj, 1.0f);
  athena_layernorm<<<B_ * S_, 256, 0, stream>>>(proj, inQ, out);
}
